// AttHead_11330123727477
// MI455X (gfx1250) — matrix operand order repaired and checked by reading, not compiled, not run on hardware
//
#include <hip/hip_runtime.h>

// ext-vector types matching the WMMA builtin signatures
typedef _Float16 __attribute__((ext_vector_type(16))) v16h;
typedef float    __attribute__((ext_vector_type(8)))  v8f;
typedef int      __attribute__((ext_vector_type(4)))  i32x4;
typedef float    __attribute__((ext_vector_type(4)))  f32x4;

#define NN     8192
#define IN_F   128
#define OUT_F  64
#define ALPHA  0.2f
#define LOG2E  1.4426950408889634f

// ---------------------------------------------------------------------------
// Kernel 1: h = X @ W  (f32), store hT (f16, [64][8192]) + f_src/f_dst.
// One wave per row; lane l owns output columns l and l+32.
// ---------------------------------------------------------------------------
__global__ __launch_bounds__(256) void gat_prep(
    const float* __restrict__ input, const float* __restrict__ W,
    const float* __restrict__ a, _Float16* __restrict__ hT,
    float* __restrict__ f_src, float* __restrict__ f_dst) {
  const int wave = threadIdx.x >> 5;
  const int lane = threadIdx.x & 31;
  const int row  = blockIdx.x * 8 + wave;
  const float* in = input + (size_t)row * IN_F;
  float h0 = 0.f, h1 = 0.f;
  #pragma unroll 4
  for (int k = 0; k < IN_F; ++k) {
    float x = in[k];
    h0 = fmaf(x, W[k * OUT_F + lane],      h0);
    h1 = fmaf(x, W[k * OUT_F + lane + 32], h1);
  }
  // transposed f16 copy -> WMMA B-fragments become contiguous 32B loads
  hT[(size_t)lane        * NN + row] = (_Float16)h0;
  hT[(size_t)(lane + 32) * NN + row] = (_Float16)h1;
  // f_src = h . a[:64], f_dst = h . a[64:]  (wave reduction, wave32)
  float ps = h0 * a[lane]         + h1 * a[lane + 32];
  float pd = h0 * a[OUT_F + lane] + h1 * a[OUT_F + lane + 32];
  #pragma unroll
  for (int off = 16; off > 0; off >>= 1) {
    ps += __shfl_xor(ps, off, 32);
    pd += __shfl_xor(pd, off, 32);
  }
  if (lane == 0) { f_src[row] = ps; f_dst[row] = pd; }
}

// ---------------------------------------------------------------------------
// Kernel 2: M = max_j f_dst[j]  (single block)
// ---------------------------------------------------------------------------
__global__ __launch_bounds__(1024) void gat_max(
    const float* __restrict__ f_dst, float* __restrict__ Mout) {
  __shared__ float sm[1024];
  float m = -3.0e38f;
  for (int i = threadIdx.x; i < NN; i += 1024) m = fmaxf(m, f_dst[i]);
  sm[threadIdx.x] = m;
  __syncthreads();
  for (int s = 512; s > 0; s >>= 1) {
    if (threadIdx.x < s) sm[threadIdx.x] = fmaxf(sm[threadIdx.x], sm[threadIdx.x + s]);
    __syncthreads();
  }
  if (threadIdx.x == 0) Mout[0] = sm[0];
}

// ---------------------------------------------------------------------------
// Kernel 3: fused  mask + leakyrelu + exp + (P @ h) + softmax-normalize + elu.
// Block = one 16-row tile (512 blocks). 8 waves split the 8192 columns in
// 32-wide K tiles. Since m_hat_i >= true row max (leakyrelu monotone), the
// softmax is a pure streaming sum: no online rescaling, waves combine by add.
// The denominator l_i = sum_j p_ij is computed on the MATRIX pipe as
// P @ ones (one extra wmma per tile) instead of 16 serial v_add_f32.
// ---------------------------------------------------------------------------
__global__ __launch_bounds__(256) void gat_main(
    const int* __restrict__ adj, const _Float16* __restrict__ hT,
    const float* __restrict__ f_src, const float* __restrict__ f_dst,
    const float* __restrict__ Mp, float* __restrict__ out) {
  __shared__ float sAcc[16 * 64];
  __shared__ float sL[16];

  const int rt     = blockIdx.x;          // 16-row tile index
  const int wave   = threadIdx.x >> 5;
  const int lane   = threadIdx.x & 31;
  const int lane16 = lane & 15;
  const bool hi    = lane >= 16;

  for (int i = threadIdx.x; i < 16 * 64; i += 256) sAcc[i] = 0.f;
  if (threadIdx.x < 16) sL[threadIdx.x] = 0.f;
  __syncthreads();

  // A-fragment row for this lane (ISA 16-bit A layout: M = lane % 16)
  const int   row  = rt * 16 + lane16;
  const float fsrc = f_src[row];
  const float Mv   = Mp[0];
  const float tmh  = fsrc + Mv;
  const float mhat = fmaxf(tmh, ALPHA * tmh);       // upper bound of row max
  const float c0   = -mhat * LOG2E;

  // per-lane K offsets (contiguous, matching B): lane<16 -> K {0..15}; else {16..31}
  const int kb  = hi ? 16 : 0;   // A / adj k-base
  const int kb2 = hi ? 16 : 0;   // B k-base (lanes 0-15 hold K 0-15, 16-31 hold K 16-31)
  const int* adjRow = adj + (size_t)row * NN;

  // all-ones B fragment: D = P @ 1 gives every row's sum in every column
  v16h Bones;
  #pragma unroll
  for (int q = 0; q < 16; ++q) Bones[q] = (_Float16)1.0f;

  v8f acc0 = {}, acc1 = {}, acc2 = {}, acc3 = {}, accL = {};

  for (int jt = wave; jt < NN / 32; jt += 8) {
    const int j0 = jt * 32;
    // 268MB adj stream: non-temporal so it can't evict hot hT from L2
    i32x4 a0 = __builtin_nontemporal_load((const i32x4*)(adjRow + j0 + kb));
    i32x4 a1 = __builtin_nontemporal_load((const i32x4*)(adjRow + j0 + kb + 4));
    i32x4 a2 = __builtin_nontemporal_load((const i32x4*)(adjRow + j0 + kb + 8));
    i32x4 a3 = __builtin_nontemporal_load((const i32x4*)(adjRow + j0 + kb + 12));
    f32x4 d0 = *(const f32x4*)(f_dst + j0 + kb);
    f32x4 d1 = *(const f32x4*)(f_dst + j0 + kb + 4);
    f32x4 d2 = *(const f32x4*)(f_dst + j0 + kb + 8);
    f32x4 d3 = *(const f32x4*)(f_dst + j0 + kb + 12);

    int av[16]; float dv[16];
    #pragma unroll
    for (int q = 0; q < 4; ++q) {
      av[q] = a0[q]; av[4 + q] = a1[q]; av[8 + q] = a2[q]; av[12 + q] = a3[q];
      dv[q] = d0[q]; dv[4 + q] = d1[q]; dv[8 + q] = d2[q]; dv[12 + q] = d3[q];
    }

    // p = mask * exp(leakyrelu(fsrc+fdst) - mhat), packed into the f16 A-fragment
    v16h A;
    #pragma unroll
    for (int q = 0; q < 16; ++q) {
      float t  = fsrc + dv[q];
      float lr = fmaxf(t, ALPHA * t);                       // leaky relu
      float ex = __builtin_exp2f(fmaf(lr, LOG2E, c0));      // exp(lr - mhat)
      float pv = (av[q] != 0) ? ex : 0.f;                   // adjacency mask
      A[q] = (_Float16)pv;
    }

    // B-fragments: hT[col][j0+kb2 .. +15] is contiguous (32B) per lane
    const _Float16* hTj = hT + j0 + kb2;
    v16h B0 = *(const v16h*)(hTj + (size_t)(lane16)      * NN);
    v16h B1 = *(const v16h*)(hTj + (size_t)(lane16 + 16) * NN);
    v16h B2 = *(const v16h*)(hTj + (size_t)(lane16 + 32) * NN);
    v16h B3 = *(const v16h*)(hTj + (size_t)(lane16 + 48) * NN);

    acc0 = __builtin_amdgcn_wmma_f32_16x16x32_f16(false, A, false, B0,    (short)0, acc0, false, false);
    acc1 = __builtin_amdgcn_wmma_f32_16x16x32_f16(false, A, false, B1,    (short)0, acc1, false, false);
    acc2 = __builtin_amdgcn_wmma_f32_16x16x32_f16(false, A, false, B2,    (short)0, acc2, false, false);
    acc3 = __builtin_amdgcn_wmma_f32_16x16x32_f16(false, A, false, B3,    (short)0, acc3, false, false);
    accL = __builtin_amdgcn_wmma_f32_16x16x32_f16(false, A, false, Bones, (short)0, accL, false, false);
  }

  // combine waves: pure sums (common shift mhat) -> LDS float atomics
  // accL: every column holds the row sum; only column-0 lanes (0 and 16) push it
  if (lane16 == 0) {
    #pragma unroll
    for (int v = 0; v < 8; ++v)
      atomicAdd(&sL[v + (hi ? 8 : 0)], accL[v]);
  }
  #pragma unroll
  for (int v = 0; v < 8; ++v) {
    const int r = v + (hi ? 8 : 0);     // ISA f32 C/D layout: M = vgpr + 8*(lane>=16)
    atomicAdd(&sAcc[r * 64 + lane16 +  0], acc0[v]);
    atomicAdd(&sAcc[r * 64 + lane16 + 16], acc1[v]);
    atomicAdd(&sAcc[r * 64 + lane16 + 32], acc2[v]);
    atomicAdd(&sAcc[r * 64 + lane16 + 48], acc3[v]);
  }
  __syncthreads();

  // normalize + elu, write output tile
  for (int idx = threadIdx.x; idx < 16 * 64; idx += 256) {
    const int r = idx >> 6, c = idx & 63;
    float x = sAcc[idx] / sL[r];
    float e = (x > 0.f) ? x : (__builtin_exp2f(x * LOG2E) - 1.f);
    out[(size_t)(rt * 16 + r) * OUT_F + c] = e;
  }
}

// ---------------------------------------------------------------------------
extern "C" void kernel_launch(void* const* d_in, const int* in_sizes, int n_in,
                              void* d_out, int out_size, void* d_ws, size_t ws_size,
                              hipStream_t stream) {
  const float* input = (const float*)d_in[0];
  const int*   adj   = (const int*)d_in[1];
  const float* W     = (const float*)d_in[2];
  const float* a     = (const float*)d_in[3];
  float* out = (float*)d_out;

  char* ws = (char*)d_ws;
  _Float16* hT    = (_Float16*)ws;                                   // 64*8192*2 = 1 MB
  float*    f_src = (float*)(ws + (size_t)OUT_F * NN * 2);           // 32 KB
  float*    f_dst = (float*)(ws + (size_t)OUT_F * NN * 2 + NN * 4);  // 32 KB
  float*    Mp    = (float*)(ws + (size_t)OUT_F * NN * 2 + 2 * NN * 4);

  gat_prep<<<NN / 8, 256, 0, stream>>>(input, W, a, hT, f_src, f_dst);
  gat_max<<<1, 1024, 0, stream>>>(f_dst, Mp);
  gat_main<<<NN / 16, 256, 0, stream>>>(adj, hT, f_src, f_dst, Mp, out);
}
